// MultiheadSelfAttention_46394236731673
// MI455X (gfx1250) — matrix operand order repaired and checked by reading, not compiled, not run on hardware
//
#include <hip/hip_runtime.h>
#include <math.h>
#include <stdint.h>

#define D_MODEL   1024
#define NUM_HEADS 16
#define D_K       64
#define SEQ       4096

typedef __attribute__((ext_vector_type(16))) _Float16 v16h;
typedef __attribute__((ext_vector_type(8)))  _Float16 v8h;
typedef __attribute__((ext_vector_type(8)))  float    v8f;

static __device__ __forceinline__ v8f wmma_f16(v16h a, v16h b, v8f c) {
    // (neg_a, A, neg_b, B, c_mod, C, reuse_a, reuse_b) -> v_wmma_f32_16x16x32_f16
    return __builtin_amdgcn_wmma_f32_16x16x32_f16(false, a, false, b, (short)0, c,
                                                  false, false);
}

// ---------------------------------------------------------------------------
// fp32 -> f16 bulk convert (8 elements/thread, vectorized)
// ---------------------------------------------------------------------------
__global__ void cvt_f16(const float* __restrict__ src, _Float16* __restrict__ dst,
                        int n) {
    const int i = (blockIdx.x * blockDim.x + threadIdx.x) * 8;
    if (i >= n) return;
    const float4* s4 = reinterpret_cast<const float4*>(src + i);
    const float4 a = s4[0], b = s4[1];
    v8h o;
    o[0] = (_Float16)a.x; o[1] = (_Float16)a.y;
    o[2] = (_Float16)a.z; o[3] = (_Float16)a.w;
    o[4] = (_Float16)b.x; o[5] = (_Float16)b.y;
    o[6] = (_Float16)b.z; o[7] = (_Float16)b.w;
    *reinterpret_cast<v8h*>(dst + i) = o;
}

// ---------------------------------------------------------------------------
// C[M,N] = A[M,K] * W[N,K]^T, all-f16 operands, fp32 out.
// 64x64 tile per workgroup (4 waves, 16 rows each).
// B tile (64n x 32k f16 = 4KB) staged in LDS with gfx1250 async-to-LDS copies,
// double buffered; one s_wait_asynccnt + barrier per K-step.
// ---------------------------------------------------------------------------
__global__ __launch_bounds__(128) void gemm_f16(const _Float16* __restrict__ A,
                                                const _Float16* __restrict__ W,
                                                float* __restrict__ C,
                                                int M, int N, int K) {
    __shared__ _Float16 Bs[2][64][32];     // 8 KB, double buffered

    const int tid  = threadIdx.x;
    const int lid  = tid & 31;
    const int wv   = tid >> 5;
    const int m16  = lid & 15;             // A row in frag / B col / D col
    const int kh   = lid >> 4;             // K-half selector
    const int nbase = blockIdx.x * 64;
    const int mbase = blockIdx.y * 64 + wv * 16;

    // Staging map: 256 x 16B chunks per K-step, 2 per thread.
    // chunk c -> n = c>>2 (0..63), kq = c&3 (8 f16 each)
    const int n0 = tid >> 1;                  // chunks c0 = 2*tid? no: use c = tid, tid+128
    const int c0 = tid,        c1 = tid + 128;
    const int sn0 = c0 >> 2,   skq0 = c0 & 3;
    const int sn1 = c1 >> 2,   skq1 = c1 & 3;
    (void)n0;

    const unsigned long long wbase = (unsigned long long)(uintptr_t)W;
    const unsigned lds00 = (unsigned)(uintptr_t)&Bs[0][sn0][skq0 * 8];
    const unsigned lds01 = (unsigned)(uintptr_t)&Bs[1][sn0][skq0 * 8];
    const unsigned lds10 = (unsigned)(uintptr_t)&Bs[0][sn1][skq1 * 8];
    const unsigned lds11 = (unsigned)(uintptr_t)&Bs[1][sn1][skq1 * 8];

    v8f acc[4] = {};
    const _Float16* arow = A + (size_t)(mbase + m16) * K;

    // ---- stage kb = 0 into buffer 0 ----
    {
        const unsigned off0 = (unsigned)(((sn0 + nbase) * K + skq0 * 8) * 2);
        const unsigned off1 = (unsigned)(((sn1 + nbase) * K + skq1 * 8) * 2);
        asm volatile("global_load_async_to_lds_b128 %0, %1, %2"
                     :: "v"(lds00), "v"(off0), "s"(wbase) : "memory");
        asm volatile("global_load_async_to_lds_b128 %0, %1, %2"
                     :: "v"(lds10), "v"(off1), "s"(wbase) : "memory");
    }

    int buf = 0;
    for (int kb = 0; kb < K; kb += 32) {
        asm volatile("s_wait_asynccnt 0" ::: "memory");
        __syncthreads();                       // tile[buf] visible to all waves

        if (kb + 32 < K) {                     // prefetch next tile into buf^1
            const unsigned off0 =
                (unsigned)(((sn0 + nbase) * K + kb + 32 + skq0 * 8) * 2);
            const unsigned off1 =
                (unsigned)(((sn1 + nbase) * K + kb + 32 + skq1 * 8) * 2);
            const unsigned l0 = buf ? lds00 : lds01;
            const unsigned l1 = buf ? lds10 : lds11;
            asm volatile("global_load_async_to_lds_b128 %0, %1, %2"
                         :: "v"(l0), "v"(off0), "s"(wbase) : "memory");
            asm volatile("global_load_async_to_lds_b128 %0, %1, %2"
                         :: "v"(l1), "v"(off1), "s"(wbase) : "memory");
        }

        // A fragment (f16 direct): elems 0..15 -> K=kb+kh*16+e
        v16h af;
        {
            const _Float16* a0 = arow + kb + kh * 16;
#pragma unroll
            for (int e = 0; e < 8; ++e) { af[e] = a0[e]; af[8 + e] = a0[8 + e]; }
        }

        // 4 B fragments from LDS, then 4 WMMAs
        v16h bfr[4];
#pragma unroll
        for (int t = 0; t < 4; ++t) {
            const _Float16* brow = &Bs[buf][t * 16 + m16][kh * 16];
#pragma unroll
            for (int e = 0; e < 16; ++e) bfr[t][e] = brow[e];
        }
#pragma unroll
        for (int t = 0; t < 4; ++t) acc[t] = wmma_f16(af, bfr[t], acc[t]);

        buf ^= 1;
    }

    // D layout: VGPR r -> row (r + 8*kh), lane -> column m16
#pragma unroll
    for (int t = 0; t < 4; ++t) {
        const int n = nbase + t * 16 + m16;
#pragma unroll
        for (int r = 0; r < 8; ++r) {
            const int mrow = mbase + r + 8 * kh;
            C[(size_t)mrow * N + n] = acc[t][r];
        }
    }
}

// ---------------------------------------------------------------------------
// RoPE on Q,K + fp32->f16 pack:
//   Qh,Kh : [head][seq][64]   (A-frag / scores-B friendly)
//   Vt    : [head][64][seq]   (V^T: PV B-frag contiguous in seq)
// ---------------------------------------------------------------------------
__global__ void rope_pack(const float* __restrict__ Qf,
                          const float* __restrict__ Kf,
                          const float* __restrict__ Vf,
                          const int*   __restrict__ pos,
                          _Float16* __restrict__ Qh,
                          _Float16* __restrict__ Kh,
                          _Float16* __restrict__ Vt) {
    const int idx = blockIdx.x * blockDim.x + threadIdx.x;
    if (idx >= SEQ * NUM_HEADS * (D_K / 2)) return;
    const int i = idx & 31;
    const int h = (idx >> 5) & 15;
    const int s = idx >> 9;

    const float p   = (float)pos[s];
    const float inv = __powf(10000.0f, -(2.0f * (float)i) / 64.0f);
    float sn, cs;
    __sincosf(p * inv, &sn, &cs);

    const size_t src = (size_t)s * D_MODEL + (size_t)h * D_K + 2 * i;
    const size_t dst = ((size_t)h * SEQ + s) * D_K + 2 * i;

    const float q1 = Qf[src], q2 = Qf[src + 1];
    Qh[dst]     = (_Float16)(q1 * cs - q2 * sn);
    Qh[dst + 1] = (_Float16)(q1 * sn + q2 * cs);

    const float k1 = Kf[src], k2 = Kf[src + 1];
    Kh[dst]     = (_Float16)(k1 * cs - k2 * sn);
    Kh[dst + 1] = (_Float16)(k1 * sn + k2 * cs);

    const float v1 = Vf[src], v2 = Vf[src + 1];
    Vt[((size_t)h * D_K + 2 * i)     * SEQ + s] = (_Float16)v1;
    Vt[((size_t)h * D_K + 2 * i + 1) * SEQ + s] = (_Float16)v2;
}

// ---------------------------------------------------------------------------
// Causal flash attention. Grid: (head, seq/64). Block: 128 (4 waves).
// Each wave owns 16 query rows; online softmax over 32-key blocks.
// Output written directly in f16 (feeds the Wo GEMM).
// ---------------------------------------------------------------------------
__global__ __launch_bounds__(128) void flash_attn(const _Float16* __restrict__ Qh,
                                                  const _Float16* __restrict__ Kh,
                                                  const _Float16* __restrict__ Vt,
                                                  _Float16* __restrict__ O) {
    __shared__ _Float16 Pl[4][16][32];      // per-wave P staging

    const int lid = threadIdx.x & 31;
    const int wv  = threadIdx.x >> 5;
    const int n16 = lid & 15;
    const int kh  = lid >> 4;
    const int h   = blockIdx.x;
    const int qb0 = blockIdx.y * 64 + wv * 16;

    const _Float16* Qp = Qh + (size_t)h * SEQ * D_K;
    const _Float16* Kp = Kh + (size_t)h * SEQ * D_K;
    const _Float16* Vp = Vt + (size_t)h * D_K * SEQ;

    // Q A-fragments: frag f covers d in [32f, 32f+32)
    v16h qa[2];
#pragma unroll
    for (int f = 0; f < 2; ++f) {
        const _Float16* qrow = Qp + (size_t)(qb0 + n16) * D_K + f * 32 + kh * 16;
#pragma unroll
        for (int e = 0; e < 8; ++e) { qa[f][e] = qrow[e]; qa[f][8 + e] = qrow[8 + e]; }
    }

    v8f   acc[4] = {};
    float mrow[8], lrow[8];
#pragma unroll
    for (int r = 0; r < 8; ++r) { mrow[r] = -INFINITY; lrow[r] = 0.0f; }

    const int kend = qb0 + 15;
    for (int kb = 0; kb <= kend; kb += 32) {
        // ---- load all 4 K B-fragments first (overlap the loads) ----
        v16h kf[2][2];
#pragma unroll
        for (int st = 0; st < 2; ++st)
#pragma unroll
            for (int f = 0; f < 2; ++f) {
                const _Float16* krow =
                    Kp + (size_t)(kb + st * 16 + n16) * D_K + f * 32 + kh * 16;
#pragma unroll
                for (int e = 0; e < 16; ++e) kf[st][f][e] = krow[e];
            }

        // ---- S = Q K^T ----
        v8f s[2] = {};
#pragma unroll
        for (int st = 0; st < 2; ++st) {
            s[st] = wmma_f16(qa[0], kf[st][0], s[st]);
            s[st] = wmma_f16(qa[1], kf[st][1], s[st]);
        }

        // ---- online softmax (rows live in 16-lane halves) ----
#pragma unroll
        for (int r = 0; r < 8; ++r) {
            const int qg = qb0 + r + 8 * kh;
            float s0 = s[0][r] * 0.125f;            // 1/sqrt(64)
            float s1 = s[1][r] * 0.125f;
            if (kb + n16      > qg) s0 = -1.0e30f;  // causal mask
            if (kb + 16 + n16 > qg) s1 = -1.0e30f;
            float bm = fmaxf(s0, s1);
#pragma unroll
            for (int off = 1; off < 16; off <<= 1)
                bm = fmaxf(bm, __shfl_xor(bm, off, 32));
            const float nm = fmaxf(mrow[r], bm);
            const float sc = __expf(mrow[r] - nm);
            const float p0 = __expf(s0 - nm);
            const float p1 = __expf(s1 - nm);
            float rs = p0 + p1;
#pragma unroll
            for (int off = 1; off < 16; off <<= 1)
                rs += __shfl_xor(rs, off, 32);
            lrow[r] = lrow[r] * sc + rs;
            mrow[r] = nm;
#pragma unroll
            for (int t = 0; t < 4; ++t) acc[t][r] *= sc;
            Pl[wv][r + 8 * kh][n16]      = (_Float16)p0;
            Pl[wv][r + 8 * kh][16 + n16] = (_Float16)p1;
        }

        // wave-local LDS round trip (cross-lane visibility)
        __builtin_amdgcn_wave_barrier();
        asm volatile("s_wait_dscnt 0" ::: "memory");
        __builtin_amdgcn_wave_barrier();

        // ---- P as A-fragment ----
        v16h pa;
        {
            const _Float16* prow = &Pl[wv][n16][kh * 16];
#pragma unroll
            for (int e = 0; e < 8; ++e) { pa[e] = prow[e]; pa[8 + e] = prow[8 + e]; }
        }

        // ---- load all 4 V B-fragments, then O += P*V ----
        v16h vf[4];
#pragma unroll
        for (int t = 0; t < 4; ++t) {
            const _Float16* vrow = Vp + (size_t)(t * 16 + n16) * SEQ + kb + kh * 16;
#pragma unroll
            for (int e = 0; e < 16; ++e) vf[t][e] = vrow[e];
        }
#pragma unroll
        for (int t = 0; t < 4; ++t) acc[t] = wmma_f16(pa, vf[t], acc[t]);
    }

    // ---- epilogue: normalize, write f16 [seq][d_model], n = h*64 + d ----
#pragma unroll
    for (int t = 0; t < 4; ++t) {
#pragma unroll
        for (int r = 0; r < 8; ++r) {
            const int qg = qb0 + r + 8 * kh;
            O[(size_t)qg * D_MODEL + h * D_K + t * 16 + n16] =
                (_Float16)(acc[t][r] / lrow[r]);
        }
    }
}

// ---------------------------------------------------------------------------
extern "C" void kernel_launch(void* const* d_in, const int* in_sizes, int n_in,
                              void* d_out, int out_size, void* d_ws, size_t ws_size,
                              hipStream_t stream) {
    const float* x   = (const float*)d_in[0];
    const int*   pos = (const int*)  d_in[1];
    const float* Wq  = (const float*)d_in[2];
    const float* Wk  = (const float*)d_in[3];
    const float* Wv  = (const float*)d_in[4];
    const float* Wo  = (const float*)d_in[5];
    float* out = (float*)d_out;

    char* ws = (char*)d_ws;
    const size_t MB = 1024 * 1024;
    float*    Qf  = (float*)(ws + 0);          // 16 MB (f32 QKV gemm outs)
    float*    Kf  = (float*)(ws + 16 * MB);    // 16 MB
    float*    Vf  = (float*)(ws + 32 * MB);    // 16 MB
    _Float16* Qh  = (_Float16*)(ws + 48 * MB); //  8 MB
    _Float16* Kh  = (_Float16*)(ws + 56 * MB); //  8 MB
    _Float16* Vt  = (_Float16*)(ws + 64 * MB); //  8 MB
    _Float16* Xh  = (_Float16*)(ws + 72 * MB); //  8 MB
    _Float16* Wqh = (_Float16*)(ws + 80 * MB); //  2 MB each
    _Float16* Wkh = (_Float16*)(ws + 82 * MB);
    _Float16* Wvh = (_Float16*)(ws + 84 * MB);
    _Float16* Woh = (_Float16*)(ws + 86 * MB);
    _Float16* ATTh = (_Float16*)(ws + 0);      //  8 MB, aliases dead Qf

    // ---- one-time f32 -> f16 conversions ----
    const int nx = SEQ * D_MODEL, nw = D_MODEL * D_MODEL;
    cvt_f16<<<nx / (8 * 256), 256, 0, stream>>>(x,  Xh,  nx);
    cvt_f16<<<nw / (8 * 256), 256, 0, stream>>>(Wq, Wqh, nw);
    cvt_f16<<<nw / (8 * 256), 256, 0, stream>>>(Wk, Wkh, nw);
    cvt_f16<<<nw / (8 * 256), 256, 0, stream>>>(Wv, Wvh, nw);
    cvt_f16<<<nw / (8 * 256), 256, 0, stream>>>(Wo, Woh, nw);

    const dim3 gg(D_MODEL / 64, SEQ / 64);
    gemm_f16<<<gg, 128, 0, stream>>>(Xh, Wqh, Qf, SEQ, D_MODEL, D_MODEL);
    gemm_f16<<<gg, 128, 0, stream>>>(Xh, Wkh, Kf, SEQ, D_MODEL, D_MODEL);
    gemm_f16<<<gg, 128, 0, stream>>>(Xh, Wvh, Vf, SEQ, D_MODEL, D_MODEL);

    const int tot = SEQ * NUM_HEADS * (D_K / 2);
    rope_pack<<<(tot + 255) / 256, 256, 0, stream>>>(Qf, Kf, Vf, pos, Qh, Kh, Vt);

    flash_attn<<<dim3(NUM_HEADS, SEQ / 64), 128, 0, stream>>>(Qh, Kh, Vt, ATTh);

    gemm_f16<<<gg, 128, 0, stream>>>(ATTh, Woh, out, SEQ, D_MODEL, D_MODEL);
}
